// BitLinearInt2_75857712382177
// MI455X (gfx1250) — matrix operand order repaired and checked by reading, not compiled, not run on hardware
//
#include <hip/hip_runtime.h>

typedef __attribute__((ext_vector_type(16))) _Float16 v16h;
typedef __attribute__((ext_vector_type(8)))  _Float16 v8h;
typedef __attribute__((ext_vector_type(8)))  float    v8f;
typedef unsigned int u32x4 __attribute__((ext_vector_type(4)));
typedef int          i32x4 __attribute__((ext_vector_type(4)));
typedef int          i32x8 __attribute__((ext_vector_type(8)));

#define M_TOK 8192
#define N_OUT 16384
#define K_IN  4096

#define BM 128
#define BN 256
#define BK 64
#define LDT 72              // BK + 8 halves padding: 144B row stride (16B-aligned, bank spread)
#define NSTAGE (K_IN / BK)  // 64 K-stages

// ---- CDNA5 async global->LDS copy (ASYNCcnt-tracked, per-lane) ----
__device__ __forceinline__ void async_copy_b128(const void* gaddr, unsigned lds_addr) {
    asm volatile("global_load_async_to_lds_b128 %0, %1, off"
                 :: "v"(lds_addr), "v"(gaddr)
                 : "memory");
}
__device__ __forceinline__ void wait_async_zero() {
    asm volatile("s_wait_asynccnt 0x0" ::: "memory");
}
// generic pointer to LDS: low 32 bits of the flat address are the LDS byte offset
__device__ __forceinline__ unsigned lds_addr_of(const void* p) {
    return (unsigned)(unsigned long long)p;
}

// ---- CDNA5 Tensor Data Mover: one 2-D tile DMA (TENSORcnt-tracked) ----
// Moves tile_d1 rows x tile_d0 elements (2B each) from a row-major tensor with
// row stride tens_stride0 into LDS, inserting 4 DWORDs of pad after every
// 32 DWORDs (=> 144B LDS row stride == LDT halves).
__device__ __forceinline__ void tdm_load_2d(const void* gaddr, unsigned lds_addr,
                                            unsigned tens_d0, unsigned tens_d1,
                                            unsigned tens_stride0,
                                            unsigned tile_d0, unsigned tile_d1) {
    unsigned long long ga = (unsigned long long)gaddr;
    // D# group 0 (4 dwords): count=1 | lds_addr | global_addr[56:0] | type=2
    u32x4 g0;
    g0[0] = 1u;                                   // count=1, user desc, no gather
    g0[1] = lds_addr;                             // LDS byte address (bits 63:32)
    g0[2] = (unsigned)(ga & 0xFFFFFFFFu);         // global_addr[31:0]  (bits 95:64)
    g0[3] = (unsigned)((ga >> 32) & 0x01FFFFFFu)  // global_addr[56:32] (bits 120:96)
          | (2u << 30);                           // type=2 ("image")   (bits 127:126)
    // D# group 1 (8 dwords)
    i32x8 g1;
    g1[0] = (int)((0u)                  // workgroup_mask = 0 (not in cluster)
          | (1u << 16)                  // data_size = 1 -> 2 bytes
          | (1u << 20)                  // pad_enable
          | (4u << 22)                  // pad_interval: enc 4 -> every 32 DWORDs (128B row)
          | (3u << 25));                // pad_amount:   enc 3 -> 4 DWORDs (16B)
    g1[1] = (int)((tens_d0 & 0xFFFFu) << 16);                    // tensor_dim0[15:0] (bits 63:48)
    g1[2] = (int)(((tens_d0 >> 16) & 0xFFFFu)                    // tensor_dim0[31:16]
          | ((tens_d1 & 0xFFFFu) << 16));                        // tensor_dim1[15:0]
    g1[3] = (int)(((tens_d1 >> 16) & 0xFFFFu)                    // tensor_dim1[31:16]
          | ((tile_d0 & 0xFFFFu) << 16));                        // tile_dim0
    g1[4] = (int)(tile_d1 & 0xFFFFu);                            // tile_dim1 (tile_dim2=0)
    g1[5] = (int)tens_stride0;                                   // tensor_dim0_stride[31:0]
    g1[6] = 0;                                                   // stride0[47:32], stride1[15:0]
    g1[7] = 0;                                                   // stride1[47:16]
    i32x4 z4 = {0, 0, 0, 0};
#if __clang_major__ >= 23
    i32x8 z8 = {0, 0, 0, 0, 0, 0, 0, 0};
    __builtin_amdgcn_tensor_load_to_lds(g0, g1, z4, z4, z8, 0);
#else
    __builtin_amdgcn_tensor_load_to_lds(g0, g1, z4, z4, 0);
#endif
}

__global__ __launch_bounds__(256) void bitlinear_int2_wmma_f16(
    const _Float16* __restrict__ x,      // [M_TOK, K_IN] row-major fp16
    const _Float16* __restrict__ w,      // [N_OUT, K_IN] row-major fp16 (ternary values)
    const _Float16* __restrict__ gamma,  // scalar fp16
    _Float16*       __restrict__ out)    // [M_TOK, N_OUT] fp16
{
    __shared__ _Float16 As[2][BM * LDT];   // 2 x 128 x 72 halves = 36 KB
    __shared__ _Float16 Bs[2][BN * LDT];   // 2 x 256 x 72 halves = 72 KB

    const int tid    = threadIdx.x;
    const int lane   = tid & 31;
    const int wave   = tid >> 5;     // 0..7
    const int wave_m = wave >> 2;    // 0..1 : which 64 rows of the block tile
    const int wave_n = wave & 3;     // 0..3 : which 64 cols of the block tile
    const int lrow   = lane & 15;    // row/col within a 16x16 tile
    const int hsel   = lane >> 4;    // which K-half this lane holds

    const int m0 = blockIdx.y * BM;
    const int n0 = blockIdx.x * BN;

    // A-tile async staging: 128 rows x 64 halves -> 4 x b128 per thread
    const int sr = tid >> 3;         // 0..31
    const int sc = (tid & 7) * 8;    // 0,8,...,56 (halves)

    v8f acc[4][4];
#pragma unroll
    for (int im = 0; im < 4; ++im)
#pragma unroll
        for (int in = 0; in < 4; ++in)
            acc[im][in] = (v8f){0.f, 0.f, 0.f, 0.f, 0.f, 0.f, 0.f, 0.f};

    // stage kt: A via per-lane async DMA (all waves), B via one TDM op (wave 0)
    auto issue_stage = [&](int kt, int buf) {
        const int k0 = kt * BK;
#pragma unroll
        for (int p = 0; p < 4; ++p) {
            async_copy_b128(x + (size_t)(m0 + sr + p * 32) * K_IN + k0 + sc,
                            lds_addr_of(&As[buf][(sr + p * 32) * LDT + sc]));
        }
        if (wave == 0) {
            tdm_load_2d(w + (size_t)n0 * K_IN + k0,      // tile start inside w
                        lds_addr_of(&Bs[buf][0]),
                        /*tensor_d0=*/K_IN, /*tensor_d1=*/N_OUT,
                        /*stride0=*/K_IN,
                        /*tile_d0=*/BK, /*tile_d1=*/BN);
        }
    };

    auto stage_fence = [&]() {
        wait_async_zero();                               // own A-tile DMA landed
        if (wave == 0) __builtin_amdgcn_s_wait_tensorcnt((short)0);  // B-tile TDM landed
        __syncthreads();                                 // whole stage visible to all
    };

    // one K=32 step: load ALL fragments into distinct regs, then 16 back-to-back WMMAs
    auto compute_ks = [&](int buf, int ks) {
        const int kb = ks * 32;
        v16h afrag[4];
        v16h bfrag[4];
#pragma unroll
        for (int im = 0; im < 4; ++im) {
            // A layout: lane half 0 holds K {0..7,16..23}, half 1 holds K {8..15,24..31}
            const _Float16* pA = &As[buf][(wave_m * 64 + im * 16 + lrow) * LDT + kb];
            v8h lo = *(const v8h*)(pA + hsel * 8);
            v8h hi = *(const v8h*)(pA + 16 + hsel * 8);
            afrag[im] = __builtin_shufflevector(lo, hi,
                0, 1, 2, 3, 4, 5, 6, 7, 8, 9, 10, 11, 12, 13, 14, 15);
        }
#pragma unroll
        for (int in = 0; in < 4; ++in) {
            // B layout: lane half 0 holds K {0..7,16..23}, half 1 holds K {8..15,24..31}
            const _Float16* pB = &Bs[buf][(wave_n * 64 + in * 16 + lrow) * LDT + kb];
            v8h lo = *(const v8h*)(pB + hsel * 8);
            v8h hi = *(const v8h*)(pB + 16 + hsel * 8);
            bfrag[in] = __builtin_shufflevector(lo, hi,
                0, 1, 2, 3, 4, 5, 6, 7, 8, 9, 10, 11, 12, 13, 14, 15);
        }
#pragma unroll
        for (int in = 0; in < 4; ++in) {
#pragma unroll
            for (int im = 0; im < 4; ++im) {
                acc[im][in] = __builtin_amdgcn_wmma_f32_16x16x32_f16(
                    /*neg_a=*/false, afrag[im],
                    /*neg_b=*/false, bfrag[in],
                    /*c_mod=*/(short)0, acc[im][in],
                    /*reuse_a=*/false, /*reuse_b=*/false);
            }
        }
    };

    // ---- prologue: DMA stage 0 into LDS buffer 0 ----
    issue_stage(0, 0);
    stage_fence();

    // ---- main loop: double-buffered LDS, async+TDM DMA overlapped with 32 WMMAs/stage ----
    for (int kt = 0; kt < NSTAGE; ++kt) {
        const int buf = kt & 1;
        if (kt + 1 < NSTAGE) issue_stage(kt + 1, buf ^ 1);  // DMA next stage
        compute_ks(buf, 0);                                 // K-step 0 (16 WMMA)
        compute_ks(buf, 1);                                 // K-step 1 (16 WMMA)
        if (kt + 1 < NSTAGE) stage_fence();
    }

    // ---- epilogue: dequant by gamma, store fp16 ----
    const float g = (float)gamma[0];
#pragma unroll
    for (int im = 0; im < 4; ++im) {
#pragma unroll
        for (int in = 0; in < 4; ++in) {
            const int row0 = m0 + wave_m * 64 + im * 16 + hsel * 8;
            const int col  = n0 + wave_n * 64 + in * 16 + lrow;
#pragma unroll
            for (int r = 0; r < 8; ++r) {
                out[(size_t)(row0 + r) * N_OUT + col] = (_Float16)(acc[im][in][r] * g);
            }
        }
    }
}

extern "C" void kernel_launch(void* const* d_in, const int* in_sizes, int n_in,
                              void* d_out, int out_size, void* d_ws, size_t ws_size,
                              hipStream_t stream) {
    const _Float16* x     = (const _Float16*)d_in[0];
    const _Float16* w     = (const _Float16*)d_in[1];
    const _Float16* gamma = (const _Float16*)d_in[2];
    _Float16*       out   = (_Float16*)d_out;

    dim3 grid(N_OUT / BN, M_TOK / BM);   // (64, 64)
    dim3 block(256);                     // 8 waves (wave32)
    bitlinear_int2_wmma_f16<<<grid, block, 0, stream>>>(x, w, gamma, out);
}
